// Attention_30030411334107
// MI455X (gfx1250) — matrix operand order repaired and checked by reading, not compiled, not run on hardware
//
#include <hip/hip_runtime.h>
#include <cstdint>
#include <cstddef>

// Flash-attention for B=4,H=16,S=2048,D=64 (fp32 in/out) on gfx1250.
// v_wmma_f32_16x16x32_f16 for both GEMMs (wave32). Double-buffered LDS
// staging with B-operand-ordered layouts, key-permuted tiles so mask
// loads are u32 and P stores are packed, deferred row-sum reduction,
// softmax in the exp2 domain with the scale folded into Q.

typedef __attribute__((ext_vector_type(16))) _Float16 v16h;
typedef __attribute__((ext_vector_type(8)))  _Float16 v8h;
typedef __attribute__((ext_vector_type(4)))  _Float16 v4h;
typedef __attribute__((ext_vector_type(8)))  float    v8f;

#define ATT_B 4
#define ATT_H 16
#define ATT_S 2048
#define ATT_D 64
#define KT    64      // keys per iteration
#define NW    8       // waves per block
#define ROWS_PER_BLOCK (NW * 16)

__device__ __forceinline__ float half_max16(float v) {
    v = fmaxf(v, __shfl_xor(v, 1));
    v = fmaxf(v, __shfl_xor(v, 2));
    v = fmaxf(v, __shfl_xor(v, 4));
    v = fmaxf(v, __shfl_xor(v, 8));
    return v;
}
__device__ __forceinline__ float half_sum16(float v) {
    v += __shfl_xor(v, 1);
    v += __shfl_xor(v, 2);
    v += __shfl_xor(v, 4);
    v += __shfl_xor(v, 8);
    return v;
}

// K tiles in QK^T B-operand order: klsB[t][c][h][lq][e] = K[key=4*lq+t][d=32c+16h+e]
// V tiles in P*V  B-operand order: vbB[t][c][h][lq][e]  = V[key=32c+16h+e][d=16t+lq]
// Every WMMA B-operand read is a wave-contiguous 1KB block.
__shared__ __align__(32) _Float16 klsB[2][4][2][2][16][16];
__shared__ __align__(32) _Float16 vbB [2][4][2][2][16][16];
__shared__ __align__(32) _Float16 pls [NW][16][KT];   // physical-key columns

__device__ __forceinline__ void stage_tile(int buf, int kt,
                                           const float* __restrict__ Kg,
                                           const float* __restrict__ Vg,
                                           size_t bh_off, int tid) {
    const int skey = tid >> 2;          // 0..63 : physical key in tile
    const int sd0  = (tid & 3) * 16;    // 0,16,32,48 : d-range base
    const float* kp = Kg + bh_off + (size_t)(kt + skey) * ATT_D + sd0;
    const float* vp = Vg + bh_off + (size_t)(kt + skey) * ATT_D + sd0;
    float4 k0 = ((const float4*)kp)[0];
    float4 k1 = ((const float4*)kp)[1];
    float4 k2 = ((const float4*)kp)[2];
    float4 k3 = ((const float4*)kp)[3];
    float4 v0 = ((const float4*)vp)[0];
    float4 v1 = ((const float4*)vp)[1];
    float4 v2 = ((const float4*)vp)[2];
    float4 v3 = ((const float4*)vp)[3];

    // K: key -> (t = key&3, lq = key>>2); d -> (c = d>>5, h = (d>>4)&1, e = d&15)
    v16h hk;
    hk[0]  = (_Float16)k0.x; hk[1]  = (_Float16)k0.y;
    hk[2]  = (_Float16)k0.z; hk[3]  = (_Float16)k0.w;
    hk[4]  = (_Float16)k1.x; hk[5]  = (_Float16)k1.y;
    hk[6]  = (_Float16)k1.z; hk[7]  = (_Float16)k1.w;
    hk[8]  = (_Float16)k2.x; hk[9]  = (_Float16)k2.y;
    hk[10] = (_Float16)k2.z; hk[11] = (_Float16)k2.w;
    hk[12] = (_Float16)k3.x; hk[13] = (_Float16)k3.y;
    hk[14] = (_Float16)k3.z; hk[15] = (_Float16)k3.w;
    *(v16h*)&klsB[buf][skey & 3][sd0 >> 5][(sd0 >> 4) & 1][skey >> 2][0] = hk;

    // V: key -> (c = key>>5, h = (key>>4)&1, e = key&15); d -> (t = d>>4, lq = d&15)
    {
        const int vt = sd0 >> 4;
        const int vc = skey >> 5;
        const int vh = (skey >> 4) & 1;
        const int ve = skey & 15;
        _Float16* vb = &vbB[buf][vt][vc][vh][0][ve];   // stride 16 halves per lq
        vb[ 0 * 16] = (_Float16)v0.x; vb[ 1 * 16] = (_Float16)v0.y;
        vb[ 2 * 16] = (_Float16)v0.z; vb[ 3 * 16] = (_Float16)v0.w;
        vb[ 4 * 16] = (_Float16)v1.x; vb[ 5 * 16] = (_Float16)v1.y;
        vb[ 6 * 16] = (_Float16)v1.z; vb[ 7 * 16] = (_Float16)v1.w;
        vb[ 8 * 16] = (_Float16)v2.x; vb[ 9 * 16] = (_Float16)v2.y;
        vb[10 * 16] = (_Float16)v2.z; vb[11 * 16] = (_Float16)v2.w;
        vb[12 * 16] = (_Float16)v3.x; vb[13 * 16] = (_Float16)v3.y;
        vb[14 * 16] = (_Float16)v3.z; vb[15 * 16] = (_Float16)v3.w;
    }
    if (kt + KT < ATT_S) {   // prefetch tile after next -> global_prefetch_b8
        __builtin_prefetch(kp + KT * ATT_D, 0, 0);
        __builtin_prefetch(vp + KT * ATT_D, 0, 0);
    }
}

__global__ __launch_bounds__(256)
void attention_fa_kernel(const float* __restrict__ Qg,
                         const float* __restrict__ Kg,
                         const float* __restrict__ Vg,
                         const uint8_t* __restrict__ Mg,  // jnp.bool_ -> 1 byte
                         float* __restrict__ Og) {
    const int tid  = threadIdx.x;
    const int wave = tid >> 5;
    const int lane = tid & 31;
    const int half = lane >> 4;   // 0 or 1
    const int lq   = lane & 15;

    const int blocks_per_head = ATT_S / ROWS_PER_BLOCK;   // 16
    const int bh    = blockIdx.x / blocks_per_head;       // 0..63
    const int qtile = blockIdx.x % blocks_per_head;
    const int qbase = qtile * ROWS_PER_BLOCK;
    const int b     = bh / ATT_H;

    const size_t bh_off = (size_t)bh * ATT_S * ATT_D;
    // fold 1/sqrt(d) and log2(e) into Q: scores come out ready for exp2
    const float scale2 = 0.125f * 1.44269504088896340736f;

    // ---- Q tile (16 rows/wave) in A-matrix layout, 2 chunks of Kdim=32 ----
    v16h qa[2];
    {
        const float* qrow = Qg + bh_off + (size_t)(qbase + wave * 16 + lq) * ATT_D;
        #pragma unroll
        for (int c = 0; c < 2; ++c) {
            const int b0 = 32 * c + 16 * half;
            #pragma unroll
            for (int j = 0; j < 8; ++j) {
                qa[c][j]     = (_Float16)(qrow[b0 + j]      * scale2);
                qa[c][8 + j] = (_Float16)(qrow[b0 + 8 + j] * scale2);
            }
        }
    }

    // online-softmax state (log2 domain). l_run is a per-lane PARTIAL sum
    // (each lane covers 4 key columns); alpha is half-uniform so the final
    // butterfly sum in the epilogue recovers the exact row sum.
    float m_run[8], l_run[8];
    v8f acc[4];
    v8f zv = {};
    #pragma unroll
    for (int r = 0; r < 8; ++r) { m_run[r] = -1e30f; l_run[r] = 0.0f; }
    #pragma unroll
    for (int t = 0; t < 4; ++t) acc[t] = zv;

    const uint8_t* mbase = Mg + (size_t)b * ATT_S * ATT_S;
    size_t mrow_off[8];
    #pragma unroll
    for (int r = 0; r < 8; ++r)
        mrow_off[r] = (size_t)(qbase + wave * 16 + r + 8 * half) * ATT_S;

    // ---- pipelined main loop: stage kt+KT while computing kt ----
    stage_tile(0, 0, Kg, Vg, bh_off, tid);
    int buf = 0;
    for (int kt = 0; kt < ATT_S; kt += KT) {
        __syncthreads();   // staged `buf` visible; prev compute on buf^1 done
        if (kt + KT < ATT_S)
            stage_tile(buf ^ 1, kt + KT, Kg, Vg, bh_off, tid);

        const _Float16* kbp = &klsB[buf][0][0][half][lq][0];
        const _Float16* vbp = &vbB [buf][0][0][half][lq][0];

        // ---- scores: 4 key tiles (tile t col lq <-> physical key 4*lq+t) ----
        v8f s[4];
        #pragma unroll
        for (int t = 0; t < 4; ++t) {
            v16h kb0 = *(const v16h*)(kbp + t * 1024 + 0 * 512);
            v16h kb1 = *(const v16h*)(kbp + t * 1024 + 1 * 512);
            s[t] = __builtin_amdgcn_wmma_f32_16x16x32_f16(false, qa[0], false, kb0,
                                                          (short)0, zv, false, false);
            s[t] = __builtin_amdgcn_wmma_f32_16x16x32_f16(false, qa[1], false, kb1,
                                                          (short)0, s[t], false, false);
        }

        // ---- mask + online softmax (rows r+8*half; keys 4*lq+0..3) ----
        #pragma unroll
        for (int r = 0; r < 8; ++r) {
            const uint32_t mw =
                *(const uint32_t*)(mbase + mrow_off[r] + kt + 4 * lq);
            float x[4];
            #pragma unroll
            for (int t = 0; t < 4; ++t) {
                x[t] = s[t][r];
                if (((mw >> (8 * t)) & 0xffu) == 0u) x[t] = -1e9f;
            }
            float lmax = fmaxf(fmaxf(x[0], x[1]), fmaxf(x[2], x[3]));
            float rowmax = half_max16(lmax);
            float mnew   = fmaxf(m_run[r], rowmax);
            float alpha  = __builtin_amdgcn_exp2f(m_run[r] - mnew);
            v4h ph;
            float psum = 0.0f;
            #pragma unroll
            for (int t = 0; t < 4; ++t) {
                float p = __builtin_amdgcn_exp2f(x[t] - mnew);
                psum += p;
                ph[t] = (_Float16)p;
            }
            *(v4h*)&pls[wave][r + 8 * half][4 * lq] = ph;   // one b64 store
            l_run[r] = l_run[r] * alpha + psum;             // partial sum
            m_run[r] = mnew;
            #pragma unroll
            for (int t = 0; t < 4; ++t) acc[t][r] *= alpha;
        }

        // ---- reload P in A-layout (physical-key columns, wave-private) ----
        v16h pa[2];
        #pragma unroll
        for (int c = 0; c < 2; ++c) {
            v8h plo = *(const v8h*)&pls[wave][lq][32 * c + 16 * half];
            v8h phi = *(const v8h*)&pls[wave][lq][32 * c + 16 * half + 8];
            #pragma unroll
            for (int j = 0; j < 8; ++j) { pa[c][j] = plo[j]; pa[c][8 + j] = phi[j]; }
        }

        // ---- P (16x64) x V (64x64): 4 d-tiles x 2 chained WMMAs ----
        #pragma unroll
        for (int t = 0; t < 4; ++t) {
            v16h vb0 = *(const v16h*)(vbp + t * 1024 + 0 * 512);
            v16h vb1 = *(const v16h*)(vbp + t * 1024 + 1 * 512);
            acc[t] = __builtin_amdgcn_wmma_f32_16x16x32_f16(false, pa[0], false, vb0,
                                                            (short)0, acc[t], false, false);
            acc[t] = __builtin_amdgcn_wmma_f32_16x16x32_f16(false, pa[1], false, vb1,
                                                            (short)0, acc[t], false, false);
        }
        buf ^= 1;
    }

    // ---- epilogue: finish row sums (one butterfly total), O = acc / l ----
    float* orow = Og + bh_off + (size_t)(qbase + wave * 16) * ATT_D;
    #pragma unroll
    for (int r = 0; r < 8; ++r) {
        float rinv = 1.0f / half_sum16(l_run[r]);
        const int row = r + 8 * half;
        #pragma unroll
        for (int t = 0; t < 4; ++t)
            orow[(size_t)row * ATT_D + 16 * t + lq] = acc[t][r] * rinv;
    }
}

extern "C" void kernel_launch(void* const* d_in, const int* in_sizes, int n_in,
                              void* d_out, int out_size, void* d_ws, size_t ws_size,
                              hipStream_t stream) {
    const float*   Q = (const float*)d_in[0];
    const float*   K = (const float*)d_in[1];
    const float*   V = (const float*)d_in[2];
    const uint8_t* M = (const uint8_t*)d_in[3];
    float*         O = (float*)d_out;

    const int blocks = ATT_B * ATT_H * (ATT_S / ROWS_PER_BLOCK);  // 1024
    attention_fa_kernel<<<dim3(blocks), dim3(256), 0, stream>>>(Q, K, V, M, O);
}
